// SparseResnetECPos_80135499809315
// MI455X (gfx1250) — matrix operand order repaired and checked by reading, not compiled, not run on hardware
//
#include <hip/hip_runtime.h>

// ---------------------------------------------------------------------------
// Types for CDNA5 WMMA (wave32): v_wmma_f32_16x16x32_bf16
// ---------------------------------------------------------------------------
typedef __bf16 bf16;
typedef __attribute__((ext_vector_type(16))) __bf16 v16bf;
typedef __attribute__((ext_vector_type(8)))  __bf16 v8bf;
typedef __attribute__((ext_vector_type(8)))  float  v8f;

#define NVERT 20000
#define NEDGE 240000

union Frag16 { v16bf v; v8bf h[2]; unsigned int u[8]; };

// Packed bf16 relu: one V_PK_MAX_NUM_BF16 per dword (2 elements).
__device__ __forceinline__ v8bf relu_v8(v8bf x) {
  union { v8bf b; unsigned int u[4]; } a;
  a.b = x;
#pragma unroll
  for (int i = 0; i < 4; ++i)
    asm("v_pk_max_num_bf16 %0, %1, 0" : "=v"(a.u[i]) : "v"(a.u[i]));
  return a.b;
}

// A matrix 16x32 bf16 layout (ISA 7.12.2): lane m = lane&15, half = lane>>4.
// v0..v3 hold K = half*8 .. +7, v4..v7 hold K = 16+half*8 .. +7.
__device__ __forceinline__ v16bf load_a_frag(const bf16* rowBase, int half) {
  Frag16 f;
  f.h[0] = *(const v8bf*)(rowBase + half * 8);
  f.h[1] = *(const v8bf*)(rowBase + 16 + half * 8);
  return f.v;
}

// B matrix 32x16 bf16 layout: lane n = lane&15, half = lane>>4.
// v0..v3 hold K = half*8 .. +7, v4..v7 hold K = 16+half*8 .. +7.
__device__ __forceinline__ v16bf load_b_frag(const bf16* rowBase, int half) {
  Frag16 f;
  f.h[0] = *(const v8bf*)(rowBase + half * 8);
  f.h[1] = *(const v8bf*)(rowBase + 16 + half * 8);
  return f.v;
}

__device__ __forceinline__ v8f wmma_bf16(v16bf a, v16bf b, v8f c) {
  return __builtin_amdgcn_wmma_f32_16x16x32_bf16(
      false, a, false, b, (short)0, c, false, false);
}

// Two floats -> packed bf16x2 dword.
__device__ __forceinline__ unsigned int pk_bf16(float a, float b) {
  union { bf16 h[2]; unsigned int u; } r;
  r.h[0] = (bf16)a;
  r.h[1] = (bf16)b;
  return r.u;
}

// Native CDNA5 fire-and-forget float atomic max (STOREcnt tracked).
__device__ __forceinline__ void atomicMaxF(float* addr, float val) {
  asm volatile("global_atomic_max_num_f32 %0, %1, off scope:SCOPE_DEV"
               :: "v"(addr), "v"(val) : "memory");
}

// ---------------------------------------------------------------------------
// Weight prep: f32 [Korig][128] -> bf16 transposed [128][Kpad], zero padded.
// ---------------------------------------------------------------------------
__global__ void k_prep_w(const float* __restrict__ src, unsigned short* __restrict__ dstT,
                         int Korig, int Kpad, int Forig, int Fpad) {
  int i = blockIdx.x * blockDim.x + threadIdx.x;
  int total = 128 * Kpad;
  if (i >= total) return;
  int n = i / Kpad;
  int k = i - n * Kpad;
  int o;
  if (Fpad > 0) {
    if (k < Fpad) o = (k < Forig) ? k : -1;
    else { int kk = k - Fpad; o = (kk < Forig) ? (Forig + kk) : -1; }
  } else {
    o = (k < Korig) ? k : -1;
  }
  float v = (o >= 0) ? src[(size_t)o * 128 + n] : 0.0f;
  reinterpret_cast<bf16*>(dstT)[(size_t)n * Kpad + k] = (bf16)v;
}

// net0[N][256] = verts @ fc_pos_w[3][256] + b
__global__ void k_fcpos(const float* __restrict__ verts, const float* __restrict__ w,
                        const float* __restrict__ b, float* __restrict__ out) {
  int i = blockIdx.x * blockDim.x + threadIdx.x;
  int node = i >> 8; int c = i & 255;
  const float* vp = verts + node * 3;
  out[i] = b[c] + vp[0] * w[c] + vp[1] * w[256 + c] + vp[2] * w[512 + c];
}

__global__ void k_init_neg_inf(float* __restrict__ p, int n) {
  int i = blockIdx.x * blockDim.x + threadIdx.x;
  if (i < n) p[i] = -__builtin_inff();
}

// agg[N][128] -> netPad[N][272] = [fixed(128), pooled(128), p(3), zeros(13)]
__global__ __launch_bounds__(128) void k_fix_build(const float* __restrict__ agg,
                                                   const float* __restrict__ verts,
                                                   float* __restrict__ netPad) {
  const int node = blockIdx.x;
  const int t = threadIdx.x;
  float v = agg[(size_t)node * 128 + t];
  if (v < -3.0e38f) v = 0.0f;           // empty segments (-inf) -> 0
  float m = v;
#pragma unroll
  for (int off = 16; off >= 1; off >>= 1) m = fmaxf(m, __shfl_xor(m, off, 32));
  __shared__ float red[4];
  if ((t & 31) == 0) red[t >> 5] = m;
  __syncthreads();
  const float pooled = fmaxf(fmaxf(red[0], red[1]), fmaxf(red[2], red[3]));
  float* row = netPad + (size_t)node * 272;
  row[t] = v;
  row[128 + t] = pooled;
  if (t < 16) row[256 + t] = (t < 3) ? verts[node * 3 + t] : 0.0f;
}

// out[N][128] = relu(agg) @ fc_c_w[128][128] + b
__global__ void k_head(const float* __restrict__ agg, const float* __restrict__ w,
                       const float* __restrict__ b, float* __restrict__ out) {
  int i = blockIdx.x * blockDim.x + threadIdx.x;
  int node = i >> 7; int c = i & 127;
  float acc = b[c];
  const float* row = agg + (size_t)node * 128;
#pragma unroll 4
  for (int k = 0; k < 128; ++k)
    acc += fmaxf(row[k], 0.0f) * w[k * 128 + c];
  out[i] = acc;
}

// ---------------------------------------------------------------------------
// Persistent EdgeConv ResNet block with WMMA.
// 256 threads / 8 waves per workgroup. w0T and wsT are cached whole in LDS
// (padded row stride -> conflict-free ds_load_b128); the workgroup loops over
// 32-edge tiles:
//   feat = [xi, xj-xi]                  (K_IN = 2*F_PAD, bf16 in LDS, raw)
//   h    = relu(feat) @ w0 + b0         (relu applied to A frags in VGPRs,
//   msg  = relu(h) @ w1 + feat @ ws + b1     v_pk_max_num_bf16)
//   agg[src[e]][:] max= msg             (native f32 atomic max)
// Wave w owns N-tile [16w,16w+16) and both M-tiles. GEMM1 (w0) and GEMM3 (ws)
// run fused in one K loop, sharing A fragments: 4 independent WMMA chains.
// ---------------------------------------------------------------------------
template <int F_PAD>
__global__ __launch_bounds__(256) void k_edgeconv(
    const float* __restrict__ nodef,                 // [N][F_PAD] f32
    const int* __restrict__ src, const int* __restrict__ dst,
    const unsigned short* __restrict__ w0T_u,        // bf16 [128][K_IN]
    const unsigned short* __restrict__ w1T_u,        // bf16 [128][128]
    const unsigned short* __restrict__ wsT_u,        // bf16 [128][K_IN]
    const float* __restrict__ b0v, const float* __restrict__ b1v,
    float* __restrict__ agg, int numTiles)
{
  constexpr int K_IN = 2 * F_PAD;
  constexpr int KSTEPS = K_IN / 32;
  constexpr int KS = K_IN + 8;   // padded LDS row stride (elements)
  constexpr int HS = 136;        // hRelu row stride (elements)

  const bf16* w0T = reinterpret_cast<const bf16*>(w0T_u);
  const bf16* w1T = reinterpret_cast<const bf16*>(w1T_u);
  const bf16* wsT = reinterpret_cast<const bf16*>(wsT_u);

  __shared__ alignas(32) bf16 sW0[128 * KS];       // 141,312 B (F_PAD=272)
  __shared__ alignas(32) bf16 sWs[128 * KS];       // 141,312 B
  __shared__ alignas(32) bf16 featRaw[32 * KS];    //  35,328 B
  __shared__ alignas(32) bf16 hRelu[32 * HS];      //   8,704 B
  __shared__ int sSrc[32];
  __shared__ int sDst[32];

  const int t    = threadIdx.x;
  const int lane = t & 31;
  const int wave = t >> 5;
  const int nloc = lane & 15;
  const int half = lane >> 4;
  const int n0   = wave * 16;            // this wave's N-tile

  // ---- Cache w0T and wsT in LDS (once per workgroup) ----
  {
    constexpr int CH = K_IN / 8;         // 16B chunks per row
    for (int i = t; i < 128 * CH; i += 256) {
      int n = i / CH, c = i - n * CH;
      const size_t go = (size_t)n * K_IN + c * 8;
      *(v8bf*)(sW0 + n * KS + c * 8) = *(const v8bf*)(w0T + go);
      *(v8bf*)(sWs + n * KS + c * 8) = *(const v8bf*)(wsT + go);
    }
  }

  const float bias0 = b0v[n0 + nloc];
  const float bias1 = b1v[n0 + nloc];
  const bf16* sW0row = sW0 + (size_t)(n0 + nloc) * KS;
  const bf16* sWsrow = sWs + (size_t)(n0 + nloc) * KS;
  const bf16* w1row  = w1T + (size_t)(n0 + nloc) * 128;
  const bf16* aRow0  = featRaw + (size_t)nloc * KS;
  const bf16* aRow1  = featRaw + (size_t)(16 + nloc) * KS;

  for (int tile = blockIdx.x; tile < numTiles; tile += gridDim.x) {
    __syncthreads();   // LDS (weights ready / previous tile fully consumed)
    const int e0 = tile * 32;
    if (t < 32) { sSrc[t] = src[e0 + t]; sDst[t] = dst[e0 + t]; }
    __syncthreads();

    // ---- Stage 32 edges of raw features into LDS as packed bf16 ----
    for (int e = wave; e < 32; e += 8) {
      const float2* xiP = (const float2*)(nodef + (size_t)sSrc[e] * F_PAD);
      const float2* xjP = (const float2*)(nodef + (size_t)sDst[e] * F_PAD);
      unsigned int* rowLo = (unsigned int*)(featRaw + e * KS);
      unsigned int* rowHi = (unsigned int*)(featRaw + e * KS + F_PAD);
      for (int fp = lane; fp < F_PAD / 2; fp += 32) {
        float2 xi = xiP[fp];
        float2 xj = xjP[fp];
        rowLo[fp] = pk_bf16(xi.x, xi.y);
        rowHi[fp] = pk_bf16(xj.x - xi.x, xj.y - xi.y);
      }
    }
    __syncthreads();

    // ---- Fused GEMM1 (h = relu(feat)@w0) + GEMM3 (shortcut = feat@ws) ----
    v8f accH0 = {}; v8f accH1 = {}; v8f accS0 = {}; v8f accS1 = {};
#pragma unroll
    for (int ks = 0; ks < KSTEPS; ++ks) {
      const int k0 = ks * 32;
      Frag16 ra0, ra1;                   // raw A fragments (2 M-tiles)
      ra0.h[0] = *(const v8bf*)(aRow0 + k0 + half * 8);
      ra0.h[1] = *(const v8bf*)(aRow0 + k0 + 16 + half * 8);
      ra1.h[0] = *(const v8bf*)(aRow1 + k0 + half * 8);
      ra1.h[1] = *(const v8bf*)(aRow1 + k0 + 16 + half * 8);
      v16bf b0f = load_b_frag(sW0row + k0, half);
      v16bf bsf = load_b_frag(sWsrow + k0, half);
      Frag16 rr0, rr1;                   // relu'd copies for the w0 path
      rr0.h[0] = relu_v8(ra0.h[0]); rr0.h[1] = relu_v8(ra0.h[1]);
      rr1.h[0] = relu_v8(ra1.h[0]); rr1.h[1] = relu_v8(ra1.h[1]);
      accH0 = wmma_bf16(rr0.v, b0f, accH0);
      accH1 = wmma_bf16(rr1.v, b0f, accH1);
      accS0 = wmma_bf16(ra0.v, bsf, accS0);
      accS1 = wmma_bf16(ra1.v, bsf, accS1);
    }
#pragma unroll
    for (int r = 0; r < 8; ++r) {
      const int m = r + 8 * half;          // C/D layout: row within M-tile
      hRelu[m * HS + n0 + nloc]        = (bf16)fmaxf(accH0[r] + bias0, 0.0f);
      hRelu[(16 + m) * HS + n0 + nloc] = (bf16)fmaxf(accH1[r] + bias0, 0.0f);
    }
    __syncthreads();

    // ---- GEMM2: msg += relu(h) @ w1 ----
#pragma unroll
    for (int ks = 0; ks < 4; ++ks) {       // K = 128 (h columns)
      const int k0 = ks * 32;
      v16bf bfrag = load_b_frag(w1row + k0, half);
      v16bf a0 = load_a_frag(hRelu + nloc * HS + k0, half);
      v16bf a1 = load_a_frag(hRelu + (16 + nloc) * HS + k0, half);
      accS0 = wmma_bf16(a0, bfrag, accS0);
      accS1 = wmma_bf16(a1, bfrag, accS1);
    }

    // ---- segment_max scatter by src (native f32 atomic max) ----
#pragma unroll
    for (int r = 0; r < 8; ++r) {
      const int m = r + 8 * half;
      atomicMaxF(agg + (size_t)sSrc[m] * 128 + n0 + nloc,      accS0[r] + bias1);
      atomicMaxF(agg + (size_t)sSrc[16 + m] * 128 + n0 + nloc, accS1[r] + bias1);
    }
  }
}

// ---------------------------------------------------------------------------
// Host launcher
// ---------------------------------------------------------------------------
extern "C" void kernel_launch(void* const* d_in, const int* in_sizes, int n_in,
                              void* d_out, int out_size, void* d_ws, size_t ws_size,
                              hipStream_t stream) {
  (void)in_sizes; (void)n_in; (void)out_size; (void)ws_size;
  const float* verts    = (const float*)d_in[0];
  const int*   eidx     = (const int*)d_in[1];
  const float* fc_pos_w = (const float*)d_in[2];
  const float* fc_pos_b = (const float*)d_in[3];
  const float* blk0_w0  = (const float*)d_in[4];
  const float* blk0_b0  = (const float*)d_in[5];
  const float* blk0_w1  = (const float*)d_in[6];
  const float* blk0_b1  = (const float*)d_in[7];
  const float* blk0_ws  = (const float*)d_in[8];
  const float* blks_w0  = (const float*)d_in[9];
  const float* blks_b0  = (const float*)d_in[10];
  const float* blks_w1  = (const float*)d_in[11];
  const float* blks_b1  = (const float*)d_in[12];
  const float* blks_ws  = (const float*)d_in[13];
  const float* fc_c_w   = (const float*)d_in[14];
  const float* fc_c_b   = (const float*)d_in[15];
  float* out = (float*)d_out;

  const int N = NVERT, E = NEDGE;
  const int* srcI = eidx;
  const int* dstI = eidx + E;

  char* ws = (char*)d_ws;
  size_t off = 0;
  auto carve = [&](size_t bytes) -> void* {
    void* p = ws + off;
    off += (bytes + 255) & ~(size_t)255;
    return p;
  };
  float* net0   = (float*)carve((size_t)N * 256 * 4);
  float* netPad = (float*)carve((size_t)N * 272 * 4);
  float* agg    = (float*)carve((size_t)N * 128 * 4);
  unsigned short* w0T0 = (unsigned short*)carve((size_t)128 * 512 * 2);
  unsigned short* wsT0 = (unsigned short*)carve((size_t)128 * 512 * 2);
  unsigned short* w1T0 = (unsigned short*)carve((size_t)128 * 128 * 2);
  unsigned short *w0Tk[4], *wsTk[4], *w1Tk[4];
  for (int k = 0; k < 4; ++k) {
    w0Tk[k] = (unsigned short*)carve((size_t)128 * 544 * 2);
    wsTk[k] = (unsigned short*)carve((size_t)128 * 544 * 2);
    w1Tk[k] = (unsigned short*)carve((size_t)128 * 128 * 2);
  }

  dim3 b256(256);
  // Weight conversion to transposed/padded bf16
  k_prep_w<<<(128 * 512 + 255) / 256, b256, 0, stream>>>(blk0_w0, w0T0, 512, 512, 0, 0);
  k_prep_w<<<(128 * 512 + 255) / 256, b256, 0, stream>>>(blk0_ws, wsT0, 512, 512, 0, 0);
  k_prep_w<<<(128 * 128 + 255) / 256, b256, 0, stream>>>(blk0_w1, w1T0, 128, 128, 0, 0);
  for (int k = 0; k < 4; ++k) {
    k_prep_w<<<(128 * 544 + 255) / 256, b256, 0, stream>>>(
        blks_w0 + (size_t)k * 518 * 128, w0Tk[k], 518, 544, 259, 272);
    k_prep_w<<<(128 * 544 + 255) / 256, b256, 0, stream>>>(
        blks_ws + (size_t)k * 518 * 128, wsTk[k], 518, 544, 259, 272);
    k_prep_w<<<(128 * 128 + 255) / 256, b256, 0, stream>>>(
        blks_w1 + (size_t)k * 128 * 128, w1Tk[k], 128, 128, 0, 0);
  }

  // fc_pos
  k_fcpos<<<(N * 256) / 256, b256, 0, stream>>>(verts, fc_pos_w, fc_pos_b, net0);

  const int numTiles = E / 32;          // 7500
  const int grid = 500;                 // persistent workgroups, 15 tiles each

  // Block 0 (feature width 256, K=512)
  k_init_neg_inf<<<(N * 128) / 256, b256, 0, stream>>>(agg, N * 128);
  k_edgeconv<256><<<grid, 256, 0, stream>>>(net0, srcI, dstI, w0T0, w1T0, wsT0,
                                            blk0_b0, blk0_b1, agg, numTiles);
  k_fix_build<<<N, 128, 0, stream>>>(agg, verts, netPad);

  // Blocks 1..4 (feature width padded 272, K=544)
  for (int k = 0; k < 4; ++k) {
    k_init_neg_inf<<<(N * 128) / 256, b256, 0, stream>>>(agg, N * 128);
    k_edgeconv<272><<<grid, 256, 0, stream>>>(netPad, srcI, dstI,
                                              w0Tk[k], w1Tk[k], wsTk[k],
                                              blks_b0 + k * 128, blks_b1 + k * 128,
                                              agg, numTiles);
    if (k < 3) k_fix_build<<<N, 128, 0, stream>>>(agg, verts, netPad);
  }

  // Head
  k_head<<<(N * 128) / 256, b256, 0, stream>>>(agg, fc_c_w, fc_c_b, out);
}
